// Transformer_90357521973387
// MI455X (gfx1250) — matrix operand order repaired and checked by reading, not compiled, not run on hardware
//
#include <hip/hip_runtime.h>

// ---------------------------------------------------------------------------
// GPT-2 small forward pass for MI455X (gfx1250), all GEMMs via
// v_wmma_f32_16x16x32_bf16 (wave32 WMMA). Compile-only target.
// GEMM: 128x128x32 block tiles, 8 wave32 waves, 32x64 wave tiles
//       -> 8 WMMA per wave per k-step, 1.5 ds_load_b128 per WMMA.
// f32->bf16 conversions use the native __bf16 cast (hardware cvt, RNE).
// ---------------------------------------------------------------------------

typedef __attribute__((ext_vector_type(16))) __bf16 v16bf;
typedef __attribute__((ext_vector_type(8)))  float  v8f;

union FragU {
    v16bf          v;
    unsigned int   u[8];
    unsigned short s[16];
};

__device__ __forceinline__ unsigned short f2bf(float f) {
    union { __bf16 h; unsigned short u; } cv;
    cv.h = (__bf16)f;          // native v_cvt (RNE) on gfx1250
    return cv.u;
}

// A-matrix (16x32 bf16) fragment K index for VGPR-pair g, lane half.
// ISA 7.12.2: VGPR0..3 hold K=0..7 (half0) / 8..15 (half1), VGPR4..7 K=16..23/24..31.
__device__ __forceinline__ int kA_of(int g, int half) {
    return (g < 4 ? 2 * g : 2 * g + 8) + 8 * half;
}
// B-matrix (32x16 bf16) fragment K index: same order as A (K=0..7/8..15, 16..23/24..31).
__device__ __forceinline__ int kB_of(int g, int half) {
    return (g < 4 ? 2 * g : 2 * g + 8) + 8 * half;
}

__device__ __forceinline__ v8f wmma_bf16(v16bf a, v16bf b, v8f c) {
    return __builtin_amdgcn_wmma_f32_16x16x32_bf16(
        false, a, false, b, (short)0, c, false, false);
}

// ---------------------------------------------------------------------------
// Embedding: x[b,t,:] = wte[idx[b,t],:] + wpe[t,:]
// ---------------------------------------------------------------------------
__global__ __launch_bounds__(256) void embed_kernel(
    const int* __restrict__ idx, const float* __restrict__ wte,
    const float* __restrict__ wpe, float* __restrict__ x, int T, int C)
{
    int row = blockIdx.x;                 // b*T + t
    int t   = row % T;
    int tok = idx[row];
    const float* we = wte + (size_t)tok * C;
    const float* pe = wpe + (size_t)t * C;
    float* xr = x + (size_t)row * C;
    for (int c = threadIdx.x; c < C; c += 256) xr[c] = we[c] + pe[c];
}

// ---------------------------------------------------------------------------
// LayerNorm over C, writes bf16 activations (feeds WMMA GEMM).
// ---------------------------------------------------------------------------
__global__ __launch_bounds__(256) void ln_kernel(
    const float* __restrict__ x, const float* __restrict__ w,
    const float* __restrict__ bb, unsigned short* __restrict__ y, int C)
{
    int row = blockIdx.x;
    const float* xr = x + (size_t)row * C;
    float s = 0.f, s2 = 0.f;
    for (int c = threadIdx.x; c < C; c += 256) {
        float v = xr[c]; s += v; s2 += v * v;
    }
#pragma unroll
    for (int o = 16; o > 0; o >>= 1) {
        s  += __shfl_down(s, o);
        s2 += __shfl_down(s2, o);
    }
    __shared__ float sh[16];
    int wv = threadIdx.x >> 5, ln = threadIdx.x & 31;
    if (ln == 0) { sh[wv] = s; sh[8 + wv] = s2; }
    __syncthreads();
    if (threadIdx.x == 0) {
        float a = 0.f, b2 = 0.f;
        for (int i = 0; i < 8; ++i) { a += sh[i]; b2 += sh[8 + i]; }
        sh[0] = a; sh[8] = b2;
    }
    __syncthreads();
    float mean = sh[0] / C;
    float var  = sh[8] / C - mean * mean;
    float rstd = rsqrtf(var + 1e-5f);
    unsigned short* yr = y + (size_t)row * C;
    for (int c = threadIdx.x; c < C; c += 256)
        yr[c] = f2bf((xr[c] - mean) * rstd * w[c] + bb[c]);
}

// ---------------------------------------------------------------------------
// WMMA GEMM: out[M,N] = act(A_bf16[M,K] @ W_f32[N,K]^T + bias) + resid
// Block: 128x128x32, 256 threads (8 wave32). Wave tile: 32 rows x 64 cols
// (2 A-frags x 4 B-frags -> 8 WMMA per k-step).
// act: 0 = none, 1 = exact GELU. outF / outB optional f32 / bf16 outputs.
// M, K multiples of 128 / 32 (true here); N edge handled (lm_head).
// ---------------------------------------------------------------------------
#define TILE_M 128
#define TILE_N 128
#define TILE_K 32
#define LDT    40   // padded LDS row stride (ushorts); 80B keeps uint4 alignment

__global__ __launch_bounds__(256) void gemm_bf16_wmma(
    const unsigned short* __restrict__ A, const float* __restrict__ W,
    const float* __restrict__ bias, const float* __restrict__ resid,
    float* __restrict__ outF, unsigned short* __restrict__ outB,
    int M, int N, int K, int act)
{
    __shared__ unsigned short sA[TILE_M * LDT];
    __shared__ unsigned short sB[TILE_N * LDT];

    const int tid  = threadIdx.x;
    const int lane = tid & 31, wave = tid >> 5;
    const int half = lane >> 4, ln = lane & 15;
    const int mw = wave & 3;      // wave M sub-tile (32 rows)
    const int nw = wave >> 2;     // wave N sub-tile (64 cols)
    const int blockM = blockIdx.y * TILE_M;
    const int blockN = blockIdx.x * TILE_N;

    const int lr = tid >> 1;           // staging row 0..127
    const int lc = (tid & 1) * 16;     // staging col 0 or 16
    const int ngl = blockN + lr;

    v8f acc[2][4];
#pragma unroll
    for (int i = 0; i < 2; ++i)
#pragma unroll
        for (int j = 0; j < 4; ++j) acc[i][j] = (v8f){};

    for (int k0 = 0; k0 < K; k0 += TILE_K) {
        // ---- stage A tile (already bf16 in memory): 16 bf16 per thread ----
        {
            const unsigned short* src = A + (size_t)(blockM + lr) * K + (k0 + lc);
            uint4 d0 = *(const uint4*)src;
            uint4 d1 = *(const uint4*)(src + 8);
            *(uint4*)(sA + lr * LDT + lc)     = d0;
            *(uint4*)(sA + lr * LDT + lc + 8) = d1;
            __builtin_prefetch(src + TILE_K, 0, 1);
        }
        // ---- stage W tile (f32 -> bf16 via native cvt): 16 floats/thread ----
        {
            union { unsigned short s[16]; uint4 q[2]; } pk;
            if (ngl < N) {
                const float* wsrc = W + (size_t)ngl * K + (k0 + lc);
                float4 f0 = *(const float4*)wsrc;
                float4 f1 = *(const float4*)(wsrc + 4);
                float4 f2 = *(const float4*)(wsrc + 8);
                float4 f3 = *(const float4*)(wsrc + 12);
                pk.s[0]  = f2bf(f0.x); pk.s[1]  = f2bf(f0.y);
                pk.s[2]  = f2bf(f0.z); pk.s[3]  = f2bf(f0.w);
                pk.s[4]  = f2bf(f1.x); pk.s[5]  = f2bf(f1.y);
                pk.s[6]  = f2bf(f1.z); pk.s[7]  = f2bf(f1.w);
                pk.s[8]  = f2bf(f2.x); pk.s[9]  = f2bf(f2.y);
                pk.s[10] = f2bf(f2.z); pk.s[11] = f2bf(f2.w);
                pk.s[12] = f2bf(f3.x); pk.s[13] = f2bf(f3.y);
                pk.s[14] = f2bf(f3.z); pk.s[15] = f2bf(f3.w);
                __builtin_prefetch(wsrc + TILE_K, 0, 1);
            } else {
                pk.q[0] = uint4{0u, 0u, 0u, 0u};
                pk.q[1] = uint4{0u, 0u, 0u, 0u};
            }
            *(uint4*)(sB + lr * LDT + lc)     = pk.q[0];
            *(uint4*)(sB + lr * LDT + lc + 8) = pk.q[1];
        }
        __syncthreads();

        // ---- gather fragments ----
        FragU fa[2], fb[4];
#pragma unroll
        for (int i = 0; i < 2; ++i) {
            const unsigned short* ab = sA + (mw * 32 + i * 16 + ln) * LDT;
#pragma unroll
            for (int g = 0; g < 8; ++g)
                fa[i].u[g] = *(const unsigned int*)(ab + kA_of(g, half));
        }
#pragma unroll
        for (int j = 0; j < 4; ++j) {
            const unsigned short* bb = sB + (nw * 64 + j * 16 + ln) * LDT;
#pragma unroll
            for (int g = 0; g < 8; ++g)
                fb[j].u[g] = *(const unsigned int*)(bb + kB_of(g, half));
        }
        // ---- 8 WMMAs ----
#pragma unroll
        for (int i = 0; i < 2; ++i)
#pragma unroll
            for (int j = 0; j < 4; ++j)
                acc[i][j] = wmma_bf16(fa[i].v, fb[j].v, acc[i][j]);
        __syncthreads();
    }

    // ---- epilogue: bias + activation + residual + store ----
#pragma unroll
    for (int i = 0; i < 2; ++i) {
#pragma unroll
        for (int j = 0; j < 4; ++j) {
            int n = blockN + nw * 64 + j * 16 + ln;
            if (n >= N) continue;
            float bv = bias ? bias[n] : 0.0f;
#pragma unroll
            for (int r = 0; r < 8; ++r) {
                int m = blockM + mw * 32 + i * 16 + r + 8 * half;
                float v = acc[i][j][r] + bv;
                if (act == 1) v = 0.5f * v * (1.0f + erff(v * 0.70710678118f));
                size_t o = (size_t)m * N + n;
                if (resid) v += resid[o];
                if (outF) outF[o] = v;
                if (outB) outB[o] = f2bf(v);
            }
        }
    }
}

// ---------------------------------------------------------------------------
// Split qkv f32 [B*T, 3C] into bf16 q/k/v [B,H,T,D].
// ---------------------------------------------------------------------------
__global__ __launch_bounds__(256) void split_qkv(
    const float* __restrict__ qkv, unsigned short* __restrict__ q,
    unsigned short* __restrict__ k, unsigned short* __restrict__ v,
    int B, int T, int Hn, int C)
{
    size_t i = (size_t)blockIdx.x * 256 + threadIdx.x;
    size_t total = (size_t)B * T * C;
    if (i >= total) return;
    int c = (int)(i % C);
    size_t bt = i / C;
    int t = (int)(bt % T);
    int b = (int)(bt / T);
    int hh = c / 64, d = c % 64;
    size_t dst = (((size_t)(b * Hn + hh)) * T + t) * 64 + d;
    const float* src = qkv + bt * (size_t)(3 * C);
    q[dst] = f2bf(src[c]);
    k[dst] = f2bf(src[C + c]);
    v[dst] = f2bf(src[2 * C + c]);
}

// ---------------------------------------------------------------------------
// Causal flash attention, one wave per 16 q-rows per (b,h).
// QK^T and P.V via bf16 WMMA; online softmax with 16-lane shfl reductions.
// Output written bf16 into [B*T, C] layout (feeds aproj GEMM).
// ---------------------------------------------------------------------------
__global__ __launch_bounds__(32) void attn_wmma(
    const unsigned short* __restrict__ Q, const unsigned short* __restrict__ Km,
    const unsigned short* __restrict__ Vm, unsigned short* __restrict__ Obf,
    int T, int Hn, int C, float scale)
{
    const int ntiles = T / 16;
    int qt = blockIdx.x % ntiles;
    int bh = blockIdx.x / ntiles;
    int b = bh / Hn, hh = bh % Hn;
    const unsigned short* qp = Q  + (size_t)bh * T * 64;
    const unsigned short* kp = Km + (size_t)bh * T * 64;
    const unsigned short* vp = Vm + (size_t)bh * T * 64;
    int lane = threadIdx.x & 31;
    int half = lane >> 4, ln = lane & 15;
    int qbase = qt * 16;

    __shared__ unsigned short sP[16 * 18];

    // Q A-fragments (D=64 -> two K=32 steps), loaded once.
    FragU aq0, aq1;
    const unsigned short* qrow = qp + (size_t)(qbase + ln) * 64;
#pragma unroll
    for (int g = 0; g < 8; ++g) {
        int kk = kA_of(g, half);
        aq0.u[g] = *(const unsigned int*)(qrow + kk);
        aq1.u[g] = *(const unsigned int*)(qrow + 32 + kk);
    }

    v8f o0 = {}, o1 = {}, o2 = {}, o3 = {};
    float mrow[8], lrow[8];
#pragma unroll
    for (int r = 0; r < 8; ++r) { mrow[r] = -3.0e38f; lrow[r] = 0.f; }

    for (int st = 0; st <= qt; ++st) {
        int sbase = st * 16;

        // K^T B-fragments: B[d][s] = K[sbase+s][d]; contiguous along d.
        FragU bk0, bk1;
        const unsigned short* krow = kp + (size_t)(sbase + ln) * 64;
#pragma unroll
        for (int g = 0; g < 8; ++g) {
            int kk = kB_of(g, half);
            bk0.u[g] = *(const unsigned int*)(krow + kk);
            bk1.u[g] = *(const unsigned int*)(krow + 32 + kk);
        }
        v8f c = {};
        c = wmma_bf16(aq0.v, bk0.v, c);
        c = wmma_bf16(aq1.v, bk1.v, c);

        int col = sbase + ln;
#pragma unroll
        for (int r = 0; r < 8; ++r) {
            int row = qbase + r + 8 * half;
            float sc = c[r] * scale;
            if (col > row) sc = -3.0e38f;
            float mx = sc;
            mx = fmaxf(mx, __shfl_xor(mx, 1));
            mx = fmaxf(mx, __shfl_xor(mx, 2));
            mx = fmaxf(mx, __shfl_xor(mx, 4));
            mx = fmaxf(mx, __shfl_xor(mx, 8));
            float mnew  = fmaxf(mrow[r], mx);
            float alpha = __expf(mrow[r] - mnew);
            float p     = __expf(sc - mnew);
            float ps = p;
            ps += __shfl_xor(ps, 1); ps += __shfl_xor(ps, 2);
            ps += __shfl_xor(ps, 4); ps += __shfl_xor(ps, 8);
            lrow[r] = lrow[r] * alpha + ps;
            mrow[r] = mnew;
            o0[r] *= alpha; o1[r] *= alpha; o2[r] *= alpha; o3[r] *= alpha;
            sP[(r + 8 * half) * 18 + ln] = f2bf(p);
        }
        __syncthreads();

        // P as A-fragment (K 0..15 valid, upper half zero).
        FragU pa;
#pragma unroll
        for (int g = 0; g < 8; ++g)
            pa.u[g] = (g < 4)
                ? *(const unsigned int*)(sP + ln * 18 + kA_of(g, half))
                : 0u;

        // V B-fragments: B[s][d] = V[sbase+s][d]; strided in s (2x b16 loads).
#pragma unroll
        for (int dt = 0; dt < 4; ++dt) {
            FragU bv;
            int dcol = dt * 16 + ln;
#pragma unroll
            for (int g = 0; g < 8; ++g) {
                int kk = kB_of(g, half);
                unsigned int val = 0u;
                if (kk < 16) {
                    const unsigned short* p0 =
                        vp + (size_t)(sbase + kk) * 64 + dcol;
                    val = (unsigned int)p0[0] | ((unsigned int)p0[64] << 16);
                }
                bv.u[g] = val;
            }
            v8f* op = (dt == 0) ? &o0 : (dt == 1) ? &o1 : (dt == 2) ? &o2 : &o3;
            *op = wmma_bf16(pa.v, bv.v, *op);
        }
        __syncthreads();
    }

    // normalize and store bf16 into [B*T, C] (heads interleaved)
#pragma unroll
    for (int dt = 0; dt < 4; ++dt) {
        v8f oc = (dt == 0) ? o0 : (dt == 1) ? o1 : (dt == 2) ? o2 : o3;
        int dcol = dt * 16 + ln;
#pragma unroll
        for (int r = 0; r < 8; ++r) {
            int trow = qbase + r + 8 * half;
            float val = oc[r] / lrow[r];
            Obf[((size_t)b * T + trow) * C + hh * 64 + dcol] = f2bf(val);
        }
    }
}

// ---------------------------------------------------------------------------
// Host orchestration
// ---------------------------------------------------------------------------
extern "C" void kernel_launch(void* const* d_in, const int* in_sizes, int n_in,
                              void* d_out, int out_size, void* d_ws, size_t ws_size,
                              hipStream_t stream) {
    (void)in_sizes; (void)n_in; (void)out_size; (void)ws_size;

    const int*   idx     = (const int*)  d_in[0];
    const float* wte     = (const float*)d_in[1];
    const float* wpe     = (const float*)d_in[2];
    const float* ln1_w   = (const float*)d_in[3];
    const float* ln1_b   = (const float*)d_in[4];
    const float* qkv_w   = (const float*)d_in[5];
    const float* qkv_b   = (const float*)d_in[6];
    const float* aproj_w = (const float*)d_in[7];
    const float* aproj_b = (const float*)d_in[8];
    const float* ln2_w   = (const float*)d_in[9];
    const float* ln2_b   = (const float*)d_in[10];
    const float* fc_w    = (const float*)d_in[11];
    const float* fc_b    = (const float*)d_in[12];
    const float* mproj_w = (const float*)d_in[13];
    const float* mproj_b = (const float*)d_in[14];
    const float* lnf_w   = (const float*)d_in[15];
    const float* lnf_b   = (const float*)d_in[16];
    const float* lm_w    = (const float*)d_in[17];

    const int B = 2, T = 1024, C = 768, Hn = 12, Ll = 6, V = 50257;
    const int M = B * T;                         // 2048 token rows

    char* ws = (char*)d_ws;
    size_t off = 0;
    auto alloc = [&](size_t bytes) -> void* {
        void* p = ws + off;
        off = (off + bytes + 255) & ~(size_t)255;
        return p;
    };
    float*          x    = (float*)         alloc((size_t)M * C * 4);
    unsigned short* h    = (unsigned short*)alloc((size_t)M * C * 2);
    float*          qkv  = (float*)         alloc((size_t)M * 3 * C * 4);
    unsigned short* qb   = (unsigned short*)alloc((size_t)M * C * 2);
    unsigned short* kb   = (unsigned short*)alloc((size_t)M * C * 2);
    unsigned short* vb   = (unsigned short*)alloc((size_t)M * C * 2);
    unsigned short* obf  = (unsigned short*)alloc((size_t)M * C * 2);
    unsigned short* fcb  = (unsigned short*)alloc((size_t)M * 4 * C * 2);

    embed_kernel<<<M, 256, 0, stream>>>(idx, wte, wpe, x, T, C);

    const dim3 blk(256);
    for (int l = 0; l < Ll; ++l) {
        // ln1 -> h (bf16)
        ln_kernel<<<M, blk, 0, stream>>>(x, ln1_w + (size_t)l * C,
                                         ln1_b + (size_t)l * C, h, C);
        // qkv = h @ qkv_w^T + qkv_b   [M, 3C] f32
        gemm_bf16_wmma<<<dim3((3 * C) / TILE_N, M / TILE_M), blk, 0, stream>>>(
            h, qkv_w + (size_t)l * 3 * C * C, qkv_b + (size_t)l * 3 * C,
            nullptr, qkv, nullptr, M, 3 * C, C, 0);
        // split heads -> bf16 q/k/v [B,H,T,D]
        split_qkv<<<((size_t)M * C + 255) / 256, blk, 0, stream>>>(
            qkv, qb, kb, vb, B, T, Hn, C);
        // attention -> obf bf16 [M, C]
        attn_wmma<<<B * Hn * (T / 16), 32, 0, stream>>>(
            qb, kb, vb, obf, T, Hn, C, 0.125f);
        // x = x + obf @ aproj_w^T + aproj_b   (in-place residual)
        gemm_bf16_wmma<<<dim3(C / TILE_N, M / TILE_M), blk, 0, stream>>>(
            obf, aproj_w + (size_t)l * C * C, aproj_b + (size_t)l * C,
            x, x, nullptr, M, C, C, 0);
        // ln2 -> h
        ln_kernel<<<M, blk, 0, stream>>>(x, ln2_w + (size_t)l * C,
                                         ln2_b + (size_t)l * C, h, C);
        // fcb = gelu(h @ fc_w^T + fc_b)  bf16 [M, 4C]
        gemm_bf16_wmma<<<dim3((4 * C) / TILE_N, M / TILE_M), blk, 0, stream>>>(
            h, fc_w + (size_t)l * 4 * C * C, fc_b + (size_t)l * 4 * C,
            nullptr, nullptr, fcb, M, 4 * C, C, 1);
        // x = x + fcb @ mproj_w^T + mproj_b
        gemm_bf16_wmma<<<dim3(C / TILE_N, M / TILE_M), blk, 0, stream>>>(
            fcb, mproj_w + (size_t)l * C * 4 * C, mproj_b + (size_t)l * C,
            x, x, nullptr, M, C, 4 * C, 0);
    }

    // final LN -> h, then logits = h @ lm_w^T   [M, V] f32 into d_out
    ln_kernel<<<M, blk, 0, stream>>>(x, lnf_w, lnf_b, h, C);
    gemm_bf16_wmma<<<dim3((V + TILE_N - 1) / TILE_N, M / TILE_M), blk, 0, stream>>>(
        h, lm_w, nullptr, nullptr, (float*)d_out, nullptr, M, V, C, 0);
}
